// ScaledDotProductAttention_31482110279957
// MI455X (gfx1250) — matrix operand order repaired and checked by reading, not compiled, not run on hardware
//
#include <hip/hip_runtime.h>
#include <hip/hip_bf16.h>

typedef __attribute__((ext_vector_type(16))) _Float16 v16h;
typedef __attribute__((ext_vector_type(8)))  float    v8f;
typedef __attribute__((ext_vector_type(2)))  __fp16   v2fp16;

#define SEQ   2048
#define DIM   64
#define QTILE 64     // query rows per block (16 per wave)
#define KCH   64     // keys per chunk (4 score tiles, 2 WMMA K-slices)
#define NWAVE 4
#define NCHUNK (SEQ / KCH)

union ABOp { uint4 u[2]; v16h h; };

static __device__ __forceinline__ v8f wmma_f16(v16h a, v16h b, v8f c) {
  return __builtin_amdgcn_wmma_f32_16x16x32_f16(
      false, a, false, b, (short)0, c, false, false);
}

static __device__ __forceinline__ unsigned pk2(float a, float b) {
#if __has_builtin(__builtin_amdgcn_cvt_pkrtz)
  union { v2fp16 h; unsigned u; } x;
  x.h = __builtin_amdgcn_cvt_pkrtz(a, b);     // v_cvt_pk_rtz_f16_f32
  return x.u;
#else
  union { _Float16 h[2]; unsigned u; } x;
  x.h[0] = (_Float16)a; x.h[1] = (_Float16)b;
  return x.u;
#endif
}

static __device__ __forceinline__ float rmax16(float v) {
  v = fmaxf(v, __shfl_xor(v, 1, 32));
  v = fmaxf(v, __shfl_xor(v, 2, 32));
  v = fmaxf(v, __shfl_xor(v, 4, 32));
  v = fmaxf(v, __shfl_xor(v, 8, 32));
  return v;
}

// permuted key position within a chunk: tiles (0,1) interleave pairwise, then (2,3);
// lets P be stored as packed b32 while keeping A-layout reads contiguous.
static __device__ __forceinline__ int kperm(int key) {
  return ((key >> 5) << 5) + ((key & 15) << 1) + ((key >> 4) & 1);
}

// ---------------- one-time fp32 -> f16 conversion of K and V into workspace ----------------
// Kh[bh][key][dim]                     : natural layout (chunk-contiguous 8KB blocks)
// Vh[bh][chunk][dim][perm(keyin)]      : exact LDS image for the PV B-operand
__global__ __launch_bounds__(256) void conv_kv(const float* __restrict__ K,
                                               const float* __restrict__ V,
                                               _Float16* __restrict__ Kh,
                                               _Float16* __restrict__ Vh) {
  const size_t idx  = (size_t)blockIdx.x * 256 + threadIdx.x;   // one float4 each
  const size_t flat = idx * 4;
  const int dim   = (int)(flat % DIM);
  const int key   = (int)((flat / DIM) % SEQ);
  const int bh    = (int)(flat / ((size_t)SEQ * DIM));

  float4 fk = *(const float4*)(K + flat);
  *(unsigned*)(Kh + flat)     = pk2(fk.x, fk.y);
  *(unsigned*)(Kh + flat + 2) = pk2(fk.z, fk.w);

  float4 fv = *(const float4*)(V + flat);
  const int chunk = key >> 6;
  const int pos   = kperm(key & 63);
  _Float16* vb = Vh + ((size_t)(bh * NCHUNK + chunk) * DIM) * KCH;
  vb[(dim + 0) * KCH + pos] = (_Float16)fv.x;
  vb[(dim + 1) * KCH + pos] = (_Float16)fv.y;
  vb[(dim + 2) * KCH + pos] = (_Float16)fv.z;
  vb[(dim + 3) * KCH + pos] = (_Float16)fv.w;
}

// ---------------- main flash-attention kernel ----------------
template <bool PRE>
__global__ __launch_bounds__(128) void fa_causal_wmma(
    const float* __restrict__ Q, const float* __restrict__ K,
    const float* __restrict__ V, const _Float16* __restrict__ Kh,
    const _Float16* __restrict__ Vh, float* __restrict__ O) {
  __shared__ _Float16 Ks[KCH][DIM];        // K chunk [key][dim]        (B-op, QK^T)
  __shared__ _Float16 Vt[DIM][KCH];        // V chunk [dim][perm(key)]  (B-op, PV)
  __shared__ _Float16 Qs[NWAVE][16][DIM];  // Q staging (pre-scaled) per wave
  __shared__ _Float16 Ps[NWAVE][16][KCH];  // probs, [row][perm(key)] -> A-layout reads

  const int tid  = threadIdx.x;
  const int wave = tid >> 5;
  const int lane = tid & 31;
  const int n    = lane & 15;
  const int hi   = lane >> 4;

  const int q0 = blockIdx.x * QTILE;
  const int bh = blockIdx.y;
  const int qw = q0 + wave * 16;

  const float* Qb = Q + (size_t)bh * SEQ * DIM;
  float*       Ob = O + (size_t)bh * SEQ * DIM;

  // scores pre-multiplied by log2(e)/64 so softmax is raw exp2 (bare v_exp_f32)
  const float qscale = 1.4426950408889634f / 64.0f;

  // ---- Stage this wave's 16x64 Q tile to LDS as f16, scale folded in ----
  {
    const int row = lane >> 1;
    const int cb  = (lane & 1) * 32;
    const float* src = Qb + (size_t)(qw + row) * DIM + cb;
#pragma unroll
    for (int j = 0; j < 8; ++j) {
      float4 f = ((const float4*)src)[j];
      *(unsigned*)&Qs[wave][row][cb + 4 * j]     = pk2(f.x * qscale, f.y * qscale);
      *(unsigned*)&Qs[wave][row][cb + 4 * j + 2] = pk2(f.z * qscale, f.w * qscale);
    }
  }
  __syncthreads();

  // ---- Q in WMMA A-layout registers: two K-slices (d 0..31, 32..63) ----
  ABOp qa0, qa1;
  {
    const _Float16* r0 = &Qs[wave][n][hi * 16];
    qa0.u[0] = *(const uint4*)(r0);
    qa0.u[1] = *(const uint4*)(r0 + 8);
    const _Float16* r1 = &Qs[wave][n][32 + hi * 16];
    qa1.u[0] = *(const uint4*)(r1);
    qa1.u[1] = *(const uint4*)(r1 + 8);
  }

  // all-ones B operand: o_l = P x ones gives rowsum(P) replicated in every lane
  v16h ones;
#pragma unroll
  for (int i = 0; i < 16; ++i) ones[i] = (_Float16)1.0f;

  v8f o0 = {}, o1 = {}, o2 = {}, o3 = {};
  v8f ol = {};
  float m_[8];
#pragma unroll
  for (int r = 0; r < 8; ++r) m_[r] = -1.0e30f;

  const int nch = (q0 + QTILE) / KCH;

  // wave-relative LDS byte addresses (flat addr low 32 bits == LDS address)
  const unsigned ldsK = (unsigned)(uintptr_t)&Ks[0][0];
  const unsigned ldsV = (unsigned)(uintptr_t)&Vt[0][0];

  for (int kc = 0; kc < nch; ++kc) {
    const int kbase = kc * KCH;

    if (PRE) {
      // ---- Async DMA: f16 chunk images from workspace straight into LDS ----
      const char* kg = (const char*)(Kh + ((size_t)bh * SEQ + kbase) * DIM);
      const char* vg = (const char*)(Vh + ((size_t)(bh * NCHUNK + kc) * DIM) * KCH);
#pragma unroll
      for (int i = 0; i < 4; ++i) {
        const int off = (tid + i * 128) * 16;     // 8KB per matrix, b128 per lane
        asm volatile("global_load_async_to_lds_b128 %0, %1, off"
                     :: "v"(ldsK + off),
                        "v"((unsigned long long)(uintptr_t)(kg + off))
                     : "memory");
        asm volatile("global_load_async_to_lds_b128 %0, %1, off"
                     :: "v"(ldsV + off),
                        "v"((unsigned long long)(uintptr_t)(vg + off))
                     : "memory");
      }
      if (kc + 1 < nch) {
        __builtin_prefetch(kg + (size_t)KCH * DIM * 2, 0, 1);
        __builtin_prefetch(vg + (size_t)KCH * DIM * 2, 0, 1);
      }
      asm volatile("s_wait_asynccnt 0x0" ::: "memory");
    } else {
      // ---- Fallback: fp32 global -> f16 LDS with inline conversion ----
      const float* Kb = K + (size_t)bh * SEQ * DIM;
      const float* Vb = V + (size_t)bh * SEQ * DIM;
#pragma unroll
      for (int i = 0; i < 8; ++i) {
        const int f4  = tid + 128 * i;
        const int key = f4 >> 4;
        const int c4  = (f4 & 15) * 4;
        const size_t g = (size_t)(kbase + key) * DIM + c4;
        float4 fk = *(const float4*)(Kb + g);
        *(unsigned*)&Ks[key][c4]     = pk2(fk.x, fk.y);
        *(unsigned*)&Ks[key][c4 + 2] = pk2(fk.z, fk.w);
        float4 fv = *(const float4*)(Vb + g);
        const int pos = kperm(key);
        Vt[c4 + 0][pos] = (_Float16)fv.x;
        Vt[c4 + 1][pos] = (_Float16)fv.y;
        Vt[c4 + 2][pos] = (_Float16)fv.z;
        Vt[c4 + 3][pos] = (_Float16)fv.w;
      }
      if (kc + 1 < nch) {
        __builtin_prefetch(Kb + (size_t)(kbase + KCH) * DIM, 0, 1);
        __builtin_prefetch(Vb + (size_t)(kbase + KCH) * DIM, 0, 1);
      }
    }
    __syncthreads();

    if (kbase <= qw + 15) {                 // wave-uniform: EXEC stays all-ones
      // ---- S = Q K^T : four 16x16 key tiles x two 32-deep slices = 8 WMMA ----
      v8f s0 = {}, s1 = {}, s2 = {}, s3 = {};
#pragma unroll
      for (int slice = 0; slice < 2; ++slice) {
        const v16h qa = slice ? qa1.h : qa0.h;
        ABOp b0, b1, b2, b3;
        const int col = slice * 32 + hi * 16;
        const _Float16* p0 = &Ks[n][col];
        b0.u[0] = *(const uint4*)(p0);      b0.u[1] = *(const uint4*)(p0 + 8);
        const _Float16* p1 = &Ks[16 + n][col];
        b1.u[0] = *(const uint4*)(p1);      b1.u[1] = *(const uint4*)(p1 + 8);
        const _Float16* p2 = &Ks[32 + n][col];
        b2.u[0] = *(const uint4*)(p2);      b2.u[1] = *(const uint4*)(p2 + 8);
        const _Float16* p3 = &Ks[48 + n][col];
        b3.u[0] = *(const uint4*)(p3);      b3.u[1] = *(const uint4*)(p3 + 8);
        s0 = wmma_f16(qa, b0.h, s0);
        s1 = wmma_f16(qa, b1.h, s1);
        s2 = wmma_f16(qa, b2.h, s2);
        s3 = wmma_f16(qa, b3.h, s3);
      }

      // ---- Online softmax in C-layout (row M = r + 8*hi, col N = n) ----
      const bool needMask = (kbase + KCH - 1) > qw;   // wave-uniform
      if (!needMask) {
#pragma unroll
        for (int r = 0; r < 8; ++r) {
          float a0 = s0[r], a1 = s1[r], a2 = s2[r], a3 = s3[r];
          const float mx = rmax16(fmaxf(fmaxf(a0, a1), fmaxf(a2, a3)));
          const float nm = fmaxf(m_[r], mx);
          const float sc = exp2f(m_[r] - nm);
          m_[r] = nm;
          const float p0 = exp2f(a0 - nm), p1 = exp2f(a1 - nm);
          const float p2 = exp2f(a2 - nm), p3 = exp2f(a3 - nm);
          const int row = 8 * hi + r;
          *(unsigned*)&Ps[wave][row][2 * n]      = pk2(p0, p1);
          *(unsigned*)&Ps[wave][row][32 + 2 * n] = pk2(p2, p3);
          o0[r] *= sc; o1[r] *= sc; o2[r] *= sc; o3[r] *= sc; ol[r] *= sc;
        }
      } else {
#pragma unroll
        for (int r = 0; r < 8; ++r) {
          const int q  = qw + 8 * hi + r;
          const int k0 = kbase + n;
          float a0 = s0[r], a1 = s1[r], a2 = s2[r], a3 = s3[r];
          if (k0      > q) a0 = -1.0e30f;
          if (k0 + 16 > q) a1 = -1.0e30f;
          if (k0 + 32 > q) a2 = -1.0e30f;
          if (k0 + 48 > q) a3 = -1.0e30f;
          const float mx = rmax16(fmaxf(fmaxf(a0, a1), fmaxf(a2, a3)));
          const float nm = fmaxf(m_[r], mx);
          const float sc = exp2f(m_[r] - nm);
          m_[r] = nm;
          const float p0 = exp2f(a0 - nm), p1 = exp2f(a1 - nm);
          const float p2 = exp2f(a2 - nm), p3 = exp2f(a3 - nm);
          const int row = 8 * hi + r;
          *(unsigned*)&Ps[wave][row][2 * n]      = pk2(p0, p1);
          *(unsigned*)&Ps[wave][row][32 + 2 * n] = pk2(p2, p3);
          o0[r] *= sc; o1[r] *= sc; o2[r] *= sc; o3[r] *= sc; ol[r] *= sc;
        }
      }
      asm volatile("s_wait_dscnt 0" ::: "memory");

      ABOp pa0, pa1;
      {
        const _Float16* pp0 = &Ps[wave][n][hi * 16];
        pa0.u[0] = *(const uint4*)(pp0);
        pa0.u[1] = *(const uint4*)(pp0 + 8);
        const _Float16* pp1 = &Ps[wave][n][32 + hi * 16];
        pa1.u[0] = *(const uint4*)(pp1);
        pa1.u[1] = *(const uint4*)(pp1 + 8);
      }
      // ---- O += P V (8 WMMA) and l += rowsum(P) (2 WMMA) ----
#pragma unroll
      for (int slice = 0; slice < 2; ++slice) {
        const v16h pa = slice ? pa1.h : pa0.h;
        ABOp v0, v1, v2, v3;
        const int col = slice * 32 + hi * 16;
        const _Float16* w0 = &Vt[n][col];
        v0.u[0] = *(const uint4*)(w0);      v0.u[1] = *(const uint4*)(w0 + 8);
        const _Float16* w1 = &Vt[16 + n][col];
        v1.u[0] = *(const uint4*)(w1);      v1.u[1] = *(const uint4*)(w1 + 8);
        const _Float16* w2 = &Vt[32 + n][col];
        v2.u[0] = *(const uint4*)(w2);      v2.u[1] = *(const uint4*)(w2 + 8);
        const _Float16* w3 = &Vt[48 + n][col];
        v3.u[0] = *(const uint4*)(w3);      v3.u[1] = *(const uint4*)(w3 + 8);
        o0 = wmma_f16(pa, v0.h, o0);
        o1 = wmma_f16(pa, v1.h, o1);
        o2 = wmma_f16(pa, v2.h, o2);
        o3 = wmma_f16(pa, v3.h, o3);
        ol = wmma_f16(pa, ones, ol);
      }
    }
    __syncthreads();
  }

  // ---- Epilogue: normalize by row sum, store fp32 ----
#pragma unroll
  for (int r = 0; r < 8; ++r) {
    const int q = qw + 8 * hi + r;
    const float inv = 1.0f / ol[r];
    float* orow = Ob + (size_t)q * DIM + n;
    orow[0]  = o0[r] * inv;
    orow[16] = o1[r] * inv;
    orow[32] = o2[r] * inv;
    orow[48] = o3[r] * inv;
  }
}

extern "C" void kernel_launch(void* const* d_in, const int* in_sizes, int n_in,
                              void* d_out, int out_size, void* d_ws, size_t ws_size,
                              hipStream_t stream) {
  const float* Q = (const float*)d_in[0];
  const float* K = (const float*)d_in[1];
  const float* V = (const float*)d_in[2];
  float* O = (float*)d_out;
  const int BH = in_sizes[0] / (SEQ * DIM);   // 2*16 = 32
  dim3 grid(SEQ / QTILE, BH);

  const size_t halfBytes = (size_t)BH * SEQ * DIM * sizeof(_Float16);  // 8 MB
  if (ws_size >= 2 * halfBytes) {
    _Float16* Kh = (_Float16*)d_ws;
    _Float16* Vh = (_Float16*)((char*)d_ws + halfBytes);
    const int nF4 = BH * SEQ * DIM / 4;
    conv_kv<<<nF4 / 256, 256, 0, stream>>>(K, V, Kh, Vh);
    fa_causal_wmma<true><<<grid, 128, 0, stream>>>(Q, K, V, Kh, Vh, O);
  } else {
    fa_causal_wmma<false><<<grid, 128, 0, stream>>>(Q, K, V, nullptr, nullptr, O);
  }
}
